// GAT_layer_2327872274952
// MI455X (gfx1250) — compile-verified, matrix operand order checked by reading, not run on hardware
//
#include <hip/hip_runtime.h>
#include <hip/hip_bf16.h>
#include <math.h>

typedef __attribute__((ext_vector_type(16))) _Float16 v16h;
typedef __attribute__((ext_vector_type(8)))  _Float16 v8h;
typedef __attribute__((ext_vector_type(8)))  float    v8f;

#define IN_DIM 128
#define HEADS  4
#define CDIM   16
#define HID    64
#define NCOLS  192   // xl | xr | res concatenated output columns
#define KTILES 4     // 128 / 32
#define CTILES 12    // 192 / 16
#define BP_HALVES (KTILES * CTILES * 32 * 16)   // 24576 halves = 48 KB
#define SLOPE  0.2f
#define LNEPS  1e-5f

// ---------------- weight pack: f32 [128 x 64]x3 -> f16 WMMA B-fragments ----------------
// Layout matches per-lane v16h load in the GEMM:
//   frag (kt,ct), lane l, half i :  K = kt*32 + (i<8 ? 8*(l>>4)+i : 16+8*(l>>4)+(i-8))
//                                   N = ct*16 + (l&15)
__global__ void pack_B_kernel(const float* __restrict__ Wl, const float* __restrict__ Wr,
                              const float* __restrict__ Wres, _Float16* __restrict__ Bp) {
  int idx = blockIdx.x * 256 + threadIdx.x;
  if (idx >= BP_HALVES) return;
  int i    = idx & 15;
  int lane = (idx >> 4) & 31;
  int ct   = (idx >> 9) % CTILES;
  int kt   = idx / (CTILES * 512);
  int nloc = lane & 15;
  int hi   = lane >> 4;
  int K    = kt * 32 + ((i < 8) ? (8 * hi + i) : (16 + 8 * hi + (i - 8)));
  int col  = ct * 16 + nloc;
  float w;
  if (col < 64)       w = Wl[K * HID + col];
  else if (col < 128) w = Wr[K * HID + (col - 64)];
  else                w = Wres[K * HID + (col - 128)];
  Bp[idx] = (_Float16)w;
}

// ---------------- x: f32 -> f16 ----------------
__global__ void cvt_x_kernel(const float* __restrict__ x, _Float16* __restrict__ xh, long total) {
  long i = (long)blockIdx.x * 256 + threadIdx.x;
  if (i < total) xh[i] = (_Float16)x[i];
}

// ---------------- fused GEMM: [N x 128] @ [128 x 192] -> xl, xr, res (f32) ----------------
// Packed B (48 KB) is staged once per block in LDS (320 KB/WGP on CDNA5) and reused by
// all 8 waves x 4 k-steps; per-fragment dependency becomes a short ds_load_b128 pair
// instead of an L2 round trip. One wave computes one 16-row x 192-col tile.
__global__ __launch_bounds__(256)
void gemm_wmma_kernel(const _Float16* __restrict__ xh, const _Float16* __restrict__ Bp,
                      const float* __restrict__ bl, const float* __restrict__ br,
                      float* __restrict__ xl, float* __restrict__ xr, float* __restrict__ res,
                      int n) {
  __shared__ _Float16 Bs[BP_HALVES];

  // cooperative fill: 24576 halves = 3072 uint4, 12 per thread
  {
    const uint4* src = (const uint4*)Bp;
    uint4*       dst = (uint4*)Bs;
    #pragma unroll
    for (int u = 0; u < BP_HALVES / 8 / 256; ++u)
      dst[threadIdx.x + u * 256] = src[threadIdx.x + u * 256];
  }
  __syncthreads();

  int wave = threadIdx.x >> 5;
  int lane = threadIdx.x & 31;
  int tile = blockIdx.x * 8 + wave;
  if (tile * 16 >= n) return;            // wave-uniform exit after barrier: EXEC all-ones for WMMA
  int row0 = tile * 16;
  int m    = lane & 15;
  int hi   = lane >> 4;
  int row  = row0 + m;
  if (row >= n) row = n - 1;             // clamp loads; stores are guarded below

  v8f acc[CTILES] = {};

  #pragma unroll
  for (int kt = 0; kt < KTILES; ++kt) {
    union { v16h v; v8h h[2]; } a;
    const _Float16* ap = xh + (long)row * IN_DIM + kt * 32;
    a.h[0] = *(const v8h*)(ap + 8 * hi);        // K = kt*32 + 8*hi .. +7
    a.h[1] = *(const v8h*)(ap + 16 + 8 * hi);   // K = kt*32 + 16 + 8*hi .. +7
    #pragma unroll
    for (int ct = 0; ct < CTILES; ++ct) {
      v16h b = *(const v16h*)(Bs + (((kt * CTILES + ct) * 32) + lane) * 16);
      acc[ct] = __builtin_amdgcn_wmma_f32_16x16x32_f16(
          false, a.v, false, b, (short)0, acc[ct], false, false);
    }
  }

  // D layout: VGPR r, lane l -> row = r + 8*(l>>4), col = l&15
  #pragma unroll
  for (int ct = 0; ct < CTILES; ++ct) {
    int col = ct * 16 + m;
    #pragma unroll
    for (int r = 0; r < 8; ++r) {
      int rr = row0 + r + 8 * hi;
      if (rr >= n) continue;
      float v = acc[ct][r];
      if (col < 64)       xl[(long)rr * HID + col]          = v + bl[col];
      else if (col < 128) xr[(long)rr * HID + (col - 64)]   = v + br[col - 64];
      else                res[(long)rr * HID + (col - 128)] = v;
    }
  }
}

// ---------------- init out = res + bias; init softmax stats ----------------
__global__ void init_out_kernel(const float4* __restrict__ res4, const float4* __restrict__ bias4,
                                float4* __restrict__ out4, long total4) {
  long i = (long)blockIdx.x * 256 + threadIdx.x;
  if (i >= total4) return;
  float4 r = res4[i];
  float4 b = bias4[(int)(i & 15)];        // HID/4 = 16 float4 per row
  out4[i] = make_float4(r.x + b.x, r.y + b.y, r.z + b.z, r.w + b.w);
}

__global__ void init_stats_kernel(float* __restrict__ smax, float* __restrict__ ssum,
                                  float* __restrict__ stats, long nh) {
  long i = (long)blockIdx.x * 256 + threadIdx.x;
  if (i < nh) { smax[i] = -INFINITY; ssum[i] = 0.f; }
  if (i < 2)  stats[i] = 0.f;
}

// ---------------- edge phase A: GATv2 scores + segment max ----------------
__device__ __forceinline__ void atomicMaxF(float* p, float v) {
  if (v >= 0.f) atomicMax((int*)p, __float_as_int(v));
  else          atomicMin((unsigned int*)p, __float_as_uint(v));
}

__device__ __forceinline__ float lrelu(float v) { return (v > 0.f) ? v : SLOPE * v; }

__global__ void edge_score_kernel(const int* __restrict__ ei, const float* __restrict__ xl,
                                  const float* __restrict__ xr, const float* __restrict__ att,
                                  float* __restrict__ score, float* __restrict__ smax,
                                  int E, int n) {
  long t  = (long)blockIdx.x * 256 + threadIdx.x;
  long Et = (long)E + n;
  if (t >= Et * HEADS) return;
  int  h = (int)(t & 3);
  long e = t >> 2;
  int s, d;
  if (e < E) { s = ei[e]; d = ei[(long)E + e]; } else { s = d = (int)(e - E); }
  const float4* pl4 = (const float4*)(xl + (long)s * HID + h * CDIM);
  const float4* pr4 = (const float4*)(xr + (long)d * HID + h * CDIM);
  const float4* at4 = (const float4*)(att + h * CDIM);
  float sc = 0.f;
  #pragma unroll
  for (int c4 = 0; c4 < 4; ++c4) {
    float4 a4 = pl4[c4];
    float4 b4 = pr4[c4];
    float4 w4 = at4[c4];
    sc = fmaf(w4.x, lrelu(a4.x + b4.x), sc);
    sc = fmaf(w4.y, lrelu(a4.y + b4.y), sc);
    sc = fmaf(w4.z, lrelu(a4.z + b4.z), sc);
    sc = fmaf(w4.w, lrelu(a4.w + b4.w), sc);
  }
  score[e * HEADS + h] = sc;
  atomicMaxF(&smax[(long)d * HEADS + h], sc);
}

// ---------------- edge phase B: exp + segment sum ----------------
__global__ void edge_exp_kernel(const int* __restrict__ ei, const float* __restrict__ smax,
                                float* __restrict__ score, float* __restrict__ ssum,
                                int E, int n) {
  long t  = (long)blockIdx.x * 256 + threadIdx.x;
  long Et = (long)E + n;
  if (t >= Et * HEADS) return;
  int  h = (int)(t & 3);
  long e = t >> 2;
  int d;
  if (e < E) d = ei[(long)E + e]; else d = (int)(e - E);
  float ex = __expf(score[e * HEADS + h] - smax[(long)d * HEADS + h]);
  score[e * HEADS + h] = ex;
  atomicAdd(&ssum[(long)d * HEADS + h], ex);
}

// ---------------- edge phase C: weighted scatter-add of messages ----------------
__global__ void edge_scatter_kernel(const int* __restrict__ ei, const float* __restrict__ xl,
                                    const float* __restrict__ score, const float* __restrict__ ssum,
                                    float* __restrict__ out, int E, int n) {
  long t  = (long)blockIdx.x * 256 + threadIdx.x;
  long Et = (long)E + n;
  if (t >= Et * HID) return;
  int  j = (int)(t & 63);
  long e = t >> 6;
  int  h = j >> 4;
  int s, d;
  if (e < E) { s = ei[e]; d = ei[(long)E + e]; } else { s = d = (int)(e - E); }
  float alpha = score[e * HEADS + h] / (ssum[(long)d * HEADS + h] + 1e-16f);
  atomicAdd(&out[(long)d * HID + j], alpha * xl[(long)s * HID + j]);
}

// ---------------- LayerNorm (graph mode): global sum / sumsq reduce, then apply ----------------
__global__ void ln_reduce_kernel(const float* __restrict__ out, float* __restrict__ stats, long M) {
  __shared__ float sh_s[256];
  __shared__ float sh_q[256];
  long i0 = (long)blockIdx.x * 1024 + threadIdx.x;
  float s = 0.f, q = 0.f;
  #pragma unroll
  for (int u = 0; u < 4; ++u) {
    long i = i0 + (long)u * 256;
    if (i < M) { float v = out[i]; s += v; q += v * v; }
  }
  sh_s[threadIdx.x] = s; sh_q[threadIdx.x] = q;
  __syncthreads();
  for (int st = 128; st > 0; st >>= 1) {
    if ((int)threadIdx.x < st) {
      sh_s[threadIdx.x] += sh_s[threadIdx.x + st];
      sh_q[threadIdx.x] += sh_q[threadIdx.x + st];
    }
    __syncthreads();
  }
  if (threadIdx.x == 0) {
    atomicAdd(&stats[0], sh_s[0]);
    atomicAdd(&stats[1], sh_q[0]);
  }
}

__global__ void ln_apply_kernel(float* __restrict__ out, const float* __restrict__ stats,
                                const float* __restrict__ lnw, const float* __restrict__ lnb,
                                long M) {
  long i = (long)blockIdx.x * 256 + threadIdx.x;
  if (i >= M) return;
  float mean = stats[0] / (float)M;
  float var  = stats[1] / (float)M - mean * mean;
  float stdv = sqrtf(fmaxf(var, 0.f));
  int j = (int)(i & 63);
  out[i] = (out[i] - mean) / (stdv + LNEPS) * lnw[j] + lnb[j];
}

extern "C" void kernel_launch(void* const* d_in, const int* in_sizes, int n_in,
                              void* d_out, int out_size, void* d_ws, size_t ws_size,
                              hipStream_t stream) {
  const float* x    = (const float*)d_in[0];
  const int*   ei   = (const int*)  d_in[1];
  const float* Wl   = (const float*)d_in[2];
  const float* bl   = (const float*)d_in[3];
  const float* Wr   = (const float*)d_in[4];
  const float* br   = (const float*)d_in[5];
  const float* att  = (const float*)d_in[6];
  const float* bias = (const float*)d_in[7];
  const float* Wres = (const float*)d_in[8];
  const float* lnw  = (const float*)d_in[9];
  const float* lnb  = (const float*)d_in[10];
  float* out = (float*)d_out;
  (void)n_in; (void)out_size; (void)ws_size;

  const long N  = (long)in_sizes[0] / IN_DIM;
  const int  E  = in_sizes[1] / 2;
  const long Et = (long)E + N;

  // workspace carve-out (256B aligned)
  char* ws = (char*)d_ws;
  size_t off = 0;
  auto alloc = [&](size_t bytes) -> char* {
    char* p = ws + off;
    off = (off + bytes + 255) & ~(size_t)255;
    return p;
  };
  _Float16* xh    = (_Float16*)alloc((size_t)N * IN_DIM * sizeof(_Float16));
  _Float16* Bp    = (_Float16*)alloc((size_t)BP_HALVES * sizeof(_Float16));
  float*    xl    = (float*)alloc((size_t)N * HID * sizeof(float));
  float*    xr    = (float*)alloc((size_t)N * HID * sizeof(float));
  float*    resb  = (float*)alloc((size_t)N * HID * sizeof(float));
  float*    score = (float*)alloc((size_t)Et * HEADS * sizeof(float));
  float*    smax  = (float*)alloc((size_t)N * HEADS * sizeof(float));
  float*    ssum  = (float*)alloc((size_t)N * HEADS * sizeof(float));
  float*    stats = (float*)alloc(2 * sizeof(float));

  // 1. pack weights to WMMA B-fragment layout
  pack_B_kernel<<<(BP_HALVES + 255) / 256, 256, 0, stream>>>(Wl, Wr, Wres, Bp);
  // 2. x -> f16
  long xtot = N * IN_DIM;
  cvt_x_kernel<<<(unsigned)((xtot + 255) / 256), 256, 0, stream>>>(x, xh, xtot);
  // 3. fused projection GEMM (xl | xr | res) via v_wmma_f32_16x16x32_f16, B staged in LDS
  long tiles = (N + 15) / 16;
  gemm_wmma_kernel<<<(unsigned)((tiles + 7) / 8), 256, 0, stream>>>(xh, Bp, bl, br, xl, xr, resb, (int)N);
  // 4. out = res + bias ; stats init
  long nelem = N * HID;
  long n4 = nelem / 4;
  init_out_kernel<<<(unsigned)((n4 + 255) / 256), 256, 0, stream>>>(
      (const float4*)resb, (const float4*)bias, (float4*)out, n4);
  long nh = N * HEADS;
  init_stats_kernel<<<(unsigned)((nh + 255) / 256), 256, 0, stream>>>(smax, ssum, stats, nh);
  // 5. scores + segment-max
  long th = Et * HEADS;
  edge_score_kernel<<<(unsigned)((th + 255) / 256), 256, 0, stream>>>(ei, xl, xr, att, score, smax, E, (int)N);
  // 6. exp + segment-sum
  edge_exp_kernel<<<(unsigned)((th + 255) / 256), 256, 0, stream>>>(ei, smax, score, ssum, E, (int)N);
  // 7. weighted scatter-add
  long ts = Et * HID;
  edge_scatter_kernel<<<(unsigned)((ts + 255) / 256), 256, 0, stream>>>(ei, xl, score, ssum, out, E, (int)N);
  // 8-9. graph LayerNorm
  long rblocks = (nelem + 1023) / 1024;
  ln_reduce_kernel<<<(unsigned)rblocks, 256, 0, stream>>>(out, stats, nelem);
  ln_apply_kernel<<<(unsigned)((nelem + 255) / 256), 256, 0, stream>>>(out, stats, lnw, lnb, nelem);
}
